// Attention_10084583211082
// MI455X (gfx1250) — matrix operand order repaired and checked by reading, not compiled, not run on hardware
//
#include <hip/hip_runtime.h>
#include <stdint.h>

// Problem constants (from reference): B=8, N=2048, C=768, NKV=512, H=12, HD=64
#define BB   8
#define NN   2048
#define CC   768
#define NKVV 512
#define HH   12
#define HDD  64
#define QSCALE 0.125f   // 64^-0.5

typedef __bf16 bf16_t;
typedef __attribute__((ext_vector_type(16))) __bf16 v16bf;
typedef __attribute__((ext_vector_type(8)))  __bf16 v8bf;
typedef __attribute__((ext_vector_type(8)))  float  v8f;
typedef __attribute__((ext_vector_type(4)))  float  v4f;

static __device__ __forceinline__ bf16_t f2bf(float f) {
  // round-to-nearest-even fp32 -> bf16
  uint32_t u = __builtin_bit_cast(uint32_t, f);
  uint32_t r = u + 0x7fffu + ((u >> 16) & 1u);
  unsigned short h = (unsigned short)(r >> 16);
  return __builtin_bit_cast(bf16_t, h);
}

static __device__ __forceinline__ v8f zero8() {
  v8f z = {0.f, 0.f, 0.f, 0.f, 0.f, 0.f, 0.f, 0.f};
  return z;
}

static __device__ __forceinline__ v8f wmma_bf16(v16bf a, v16bf b, v8f c) {
  // (neg_a, A, neg_b, B, c_mod, C, reuse_a, reuse_b)
  return __builtin_amdgcn_wmma_f32_16x16x32_bf16(false, a, false, b, (short)0, c,
                                                 false, false);
}

// ---- fragment loaders (bf16 only; hot loops do zero format conversion) ------
// A 16x32 bf16 layout (wave32): lane<16 row=lane  : elems = k[0..7], k[16..23]
//                               lane>=16 row=ln-16: elems = k[8..15], k[24..31]
static __device__ __forceinline__ v16bf load_a_bf16(const bf16_t* __restrict__ rowp,
                                                    int k0, int hi) {
  const int o0 = k0 + (hi ? 8 : 0);
  v8bf h0 = *(const v8bf*)(rowp + o0);
  v8bf h1 = *(const v8bf*)(rowp + o0 + 16);
  return __builtin_shufflevector(h0, h1, 0, 1, 2, 3, 4, 5, 6, 7,
                                 8, 9, 10, 11, 12, 13, 14, 15);
}

// B 32x16 bf16 layout (wave32): lane<16 col=lane  : elems = k[0..7], k[16..23]
//                               lane>=16 col=ln-16: elems = k[8..15], k[24..31]
static __device__ __forceinline__ v16bf load_b_bf16(const bf16_t* __restrict__ colp,
                                                    int k0, int hi) {
  const int o = k0 + (hi ? 8 : 0);
  v8bf h0 = *(const v8bf*)(colp + o);
  v8bf h1 = *(const v8bf*)(colp + o + 16);
  return __builtin_shufflevector(h0, h1, 0, 1, 2, 3, 4, 5, 6, 7,
                                 8, 9, 10, 11, 12, 13, 14, 15);
}

// ---- kernel 0: one-shot fp32 -> bf16 conversion (x and all weights) ---------
__global__ __launch_bounds__(256) void cvt_kernel(const float* __restrict__ src,
                                                  bf16_t* __restrict__ dst) {
  const size_t i = ((size_t)blockIdx.x * 256 + threadIdx.x) * 8;
  v4f f0 = *(const v4f*)(src + i);
  v4f f1 = *(const v4f*)(src + i + 4);
  v8bf o;
#pragma unroll
  for (int k = 0; k < 4; ++k) { o[k] = f2bf(f0[k]); o[4 + k] = f2bf(f1[k]); }
  *(v8bf*)(dst + i) = o;
}

// ---- kernel 1: Q = (x @ q_w^T) * SCALE, layout Q[b][h][n][64] bf16 ----------
// one wave -> 32 rows x 64 cols; jobs = (B*N/32) * (C/64) = 512*12 = 6144
__global__ __launch_bounds__(256) void qproj_kernel(const bf16_t* __restrict__ xb,
                                                    const bf16_t* __restrict__ qw,
                                                    bf16_t* __restrict__ Q) {
  const int lane = threadIdx.x & 31, hi = lane >> 4, lr = lane & 15;
  const int job = blockIdx.x * 8 + (threadIdx.x >> 5);
  const int mt = job / (CC / HDD);
  const int h  = job % (CC / HDD);
  const int row0 = mt * 32;
  const bf16_t* arow0 = xb + (size_t)(row0 + lr) * CC;
  const bf16_t* arow1 = xb + (size_t)(row0 + 16 + lr) * CC;
  v8f acc[2][4];
#pragma unroll
  for (int i = 0; i < 2; ++i)
#pragma unroll
    for (int j = 0; j < 4; ++j) acc[i][j] = zero8();
  for (int k0 = 0; k0 < CC; k0 += 32) {
    v16bf a0 = load_a_bf16(arow0, k0, hi);
    v16bf a1 = load_a_bf16(arow1, k0, hi);
#pragma unroll
    for (int j = 0; j < 4; ++j) {
      v16bf bfr = load_b_bf16(qw + (size_t)(h * HDD + j * 16 + lr) * CC, k0, hi);
      acc[0][j] = wmma_bf16(a0, bfr, acc[0][j]);
      acc[1][j] = wmma_bf16(a1, bfr, acc[1][j]);
    }
  }
#pragma unroll
  for (int i = 0; i < 2; ++i)
#pragma unroll
    for (int j = 0; j < 4; ++j)
#pragma unroll
      for (int r = 0; r < 8; ++r) {
        const int rr = row0 + i * 16 + r + 8 * hi;
        const int b = rr >> 11, n = rr & (NN - 1);
        const int d = j * 16 + lr;
        Q[(((size_t)b * HH + h) * NN + n) * HDD + d] = f2bf(acc[i][j][r] * QSCALE);
      }
}

// ---- kernel 2: KV = gather(x) @ kv_w^T; K[b][h][m][64], Vt[b][h][d][512] ----
// one wave -> 32 m-rows x 64 cols; jobs = B * (NKV/32) * (2C/64) = 8*16*24 = 3072
__global__ __launch_bounds__(256) void kvproj_kernel(const bf16_t* __restrict__ xb,
                                                     const int* __restrict__ key_ind,
                                                     const bf16_t* __restrict__ kvw,
                                                     bf16_t* __restrict__ K,
                                                     bf16_t* __restrict__ Vt) {
  const int lane = threadIdx.x & 31, hi = lane >> 4, lr = lane & 15;
  const int job = blockIdx.x * 8 + (threadIdx.x >> 5);
  const int b  = job / (16 * 24);
  const int r2 = job % (16 * 24);
  const int mt = r2 / 24, cg = r2 % 24;
  const int m0 = mt * 32;
  const int ki0 = key_ind[b * NKVV + m0 + lr];        // per-lane gathered rows
  const int ki1 = key_ind[b * NKVV + m0 + 16 + lr];
  const bf16_t* arow0 = xb + ((size_t)b * NN + ki0) * CC;
  const bf16_t* arow1 = xb + ((size_t)b * NN + ki1) * CC;
  v8f acc[2][4];
#pragma unroll
  for (int i = 0; i < 2; ++i)
#pragma unroll
    for (int j = 0; j < 4; ++j) acc[i][j] = zero8();
  for (int k0 = 0; k0 < CC; k0 += 32) {
    v16bf a0 = load_a_bf16(arow0, k0, hi);
    v16bf a1 = load_a_bf16(arow1, k0, hi);
#pragma unroll
    for (int j = 0; j < 4; ++j) {
      v16bf bfr = load_b_bf16(kvw + (size_t)(cg * 64 + j * 16 + lr) * CC, k0, hi);
      acc[0][j] = wmma_bf16(a0, bfr, acc[0][j]);
      acc[1][j] = wmma_bf16(a1, bfr, acc[1][j]);
    }
  }
  const int h = cg >> 1;
  const bool isV = cg & 1;
#pragma unroll
  for (int i = 0; i < 2; ++i)
#pragma unroll
    for (int j = 0; j < 4; ++j)
#pragma unroll
      for (int r = 0; r < 8; ++r) {
        const int m = m0 + i * 16 + r + 8 * hi;
        const int d = j * 16 + lr;
        const bf16_t val = f2bf(acc[i][j][r]);
        if (!isV) K [(((size_t)b * HH + h) * NKVV + m) * HDD + d] = val;
        else      Vt[(((size_t)b * HH + h) * HDD + d) * NKVV + m] = val;
      }
}

// ---- kernel 3: flash attention; Ao[b][n][c] bf16 ----------------------------
// block = 8 waves, wave = 16 query rows; grid = B*H * (N/128) = 96*16 = 1536
__global__ __launch_bounds__(256) void attn_kernel(const bf16_t* __restrict__ Q,
                                                   const bf16_t* __restrict__ K,
                                                   const bf16_t* __restrict__ Vt,
                                                   bf16_t* __restrict__ Ao) {
  __shared__ __align__(16) bf16_t Pl[8][16][32];  // per-wave P transpose buffer
  const int lane = threadIdx.x & 31, hi = lane >> 4, lr = lane & 15;
  const int wave = threadIdx.x >> 5;
  const int bh = blockIdx.x >> 4;
  const int nt = blockIdx.x & 15;
  const int b = bh / HH, h = bh % HH;
  const int n0 = nt * 128 + wave * 16;
  const bf16_t* Qp = Q  + ((size_t)b * HH + h) * NN * HDD;
  const bf16_t* Kp = K  + ((size_t)b * HH + h) * NKVV * HDD;
  const bf16_t* Vp = Vt + ((size_t)b * HH + h) * HDD * NKVV;

  const bf16_t* qrow = Qp + (size_t)(n0 + lr) * HDD;
  const v16bf qa0 = load_a_bf16(qrow, 0, hi);     // d = 0..31
  const v16bf qa1 = load_a_bf16(qrow, 32, hi);    // d = 32..63

  v8f acc[4] = {zero8(), zero8(), zero8(), zero8()};
  float rmax[8], rsum[8];
#pragma unroll
  for (int r = 0; r < 8; ++r) { rmax[r] = -3.0e38f; rsum[r] = 0.f; }

  for (int m0 = 0; m0 < NKVV; m0 += 32) {
    // S = Q @ K^T for 32 keys -> two 16x16 C-fragments
    const bf16_t* k0p = Kp + (size_t)(m0 + lr) * HDD;
    const bf16_t* k1p = Kp + (size_t)(m0 + 16 + lr) * HDD;
    v8f s0 = zero8(), s1 = zero8();
    s0 = wmma_bf16(qa0, load_b_bf16(k0p, 0, hi), s0);
    s0 = wmma_bf16(qa1, load_b_bf16(k0p, 32, hi), s0);
    s1 = wmma_bf16(qa0, load_b_bf16(k1p, 0, hi), s1);
    s1 = wmma_bf16(qa1, load_b_bf16(k1p, 32, hi), s1);

    // online softmax: C-layout puts one row across the 16-lane half per VGPR
    float psc[8];
#pragma unroll
    for (int r = 0; r < 8; ++r) {
      float t = fmaxf(s0[r], s1[r]);
      t = fmaxf(t, __shfl_xor(t, 1, 32));
      t = fmaxf(t, __shfl_xor(t, 2, 32));
      t = fmaxf(t, __shfl_xor(t, 4, 32));
      t = fmaxf(t, __shfl_xor(t, 8, 32));
      const float nm = fmaxf(rmax[r], t);
      psc[r] = __expf(rmax[r] - nm);
      rmax[r] = nm;
      const float p0 = __expf(s0[r] - nm);
      const float p1 = __expf(s1[r] - nm);
      s0[r] = p0; s1[r] = p1;
      float ps = p0 + p1;
      ps += __shfl_xor(ps, 1, 32);
      ps += __shfl_xor(ps, 2, 32);
      ps += __shfl_xor(ps, 4, 32);
      ps += __shfl_xor(ps, 8, 32);
      rsum[r] = rsum[r] * psc[r] + ps;
    }
#pragma unroll
    for (int j = 0; j < 4; ++j)
#pragma unroll
      for (int r = 0; r < 8; ++r) acc[j][r] *= psc[r];

    // P: C-layout -> LDS -> A-layout (transpose within the wave)
#pragma unroll
    for (int r = 0; r < 8; ++r) {
      Pl[wave][r + 8 * hi][lr]      = f2bf(s0[r]);
      Pl[wave][r + 8 * hi][16 + lr] = f2bf(s1[r]);
    }
    asm volatile("s_wait_dscnt 0" ::: "memory");   // CDNA5 split DS counter
    v16bf pa;
    {
      const bf16_t* pr = &Pl[wave][lr][0];
      const int o0 = hi ? 8 : 0;
      v8bf h0 = *(const v8bf*)(pr + o0);
      v8bf h1 = *(const v8bf*)(pr + o0 + 16);
      pa = __builtin_shufflevector(h0, h1, 0, 1, 2, 3, 4, 5, 6, 7,
                                   8, 9, 10, 11, 12, 13, 14, 15);
    }
    // O += P @ V  (Vt gives contiguous 32B B-fragment loads)
#pragma unroll
    for (int j = 0; j < 4; ++j) {
      const bf16_t* vcol = Vp + (size_t)(j * 16 + lr) * NKVV + m0;
      v8bf h0 = *(const v8bf*)(vcol + (hi ? 8 : 0));
      v8bf h1 = *(const v8bf*)(vcol + (hi ? 24 : 16));
      v16bf vb = __builtin_shufflevector(h0, h1, 0, 1, 2, 3, 4, 5, 6, 7,
                                         8, 9, 10, 11, 12, 13, 14, 15);
      acc[j] = wmma_bf16(pa, vb, acc[j]);
    }
  }

#pragma unroll
  for (int r = 0; r < 8; ++r) rsum[r] = 1.0f / rsum[r];
#pragma unroll
  for (int j = 0; j < 4; ++j)
#pragma unroll
    for (int r = 0; r < 8; ++r) {
      const int n = n0 + r + 8 * hi;
      const int c = h * HDD + j * 16 + lr;
      Ao[((size_t)b * NN + n) * CC + c] = f2bf(acc[j][r] * rsum[r]);
    }
}

// ---- kernel 4: out = Ao @ proj_w^T + proj_b (fp32) --------------------------
// one wave -> 32 rows x 64 cols; jobs = 512*12 = 6144
__global__ __launch_bounds__(256) void oproj_kernel(const bf16_t* __restrict__ Ao,
                                                    const bf16_t* __restrict__ pw,
                                                    const float* __restrict__ pb,
                                                    float* __restrict__ out) {
  const int lane = threadIdx.x & 31, hi = lane >> 4, lr = lane & 15;
  const int job = blockIdx.x * 8 + (threadIdx.x >> 5);
  const int mt = job / (CC / HDD);
  const int cg = job % (CC / HDD);
  const int row0 = mt * 32;
  const bf16_t* arow0 = Ao + (size_t)(row0 + lr) * CC;
  const bf16_t* arow1 = Ao + (size_t)(row0 + 16 + lr) * CC;
  v8f acc[2][4];
#pragma unroll
  for (int i = 0; i < 2; ++i)
#pragma unroll
    for (int j = 0; j < 4; ++j) acc[i][j] = zero8();
  for (int k0 = 0; k0 < CC; k0 += 32) {
    v16bf a0 = load_a_bf16(arow0, k0, hi);
    v16bf a1 = load_a_bf16(arow1, k0, hi);
#pragma unroll
    for (int j = 0; j < 4; ++j) {
      v16bf bfr = load_b_bf16(pw + (size_t)(cg * 64 + j * 16 + lr) * CC, k0, hi);
      acc[0][j] = wmma_bf16(a0, bfr, acc[0][j]);
      acc[1][j] = wmma_bf16(a1, bfr, acc[1][j]);
    }
  }
#pragma unroll
  for (int j = 0; j < 4; ++j) {
    const int c = cg * 64 + j * 16 + lr;
    const float bias = pb[c];
#pragma unroll
    for (int i = 0; i < 2; ++i)
#pragma unroll
      for (int r = 0; r < 8; ++r) {
        const int rr = row0 + i * 16 + r + 8 * hi;
        out[(size_t)rr * CC + c] = acc[i][j][r] + bias;
      }
  }
}

extern "C" void kernel_launch(void* const* d_in, const int* in_sizes, int n_in,
                              void* d_out, int out_size, void* d_ws, size_t ws_size,
                              hipStream_t stream) {
  (void)in_sizes; (void)n_in; (void)out_size; (void)ws_size;
  const float* x       = (const float*)d_in[0];
  const int*   key_ind = (const int*)  d_in[1];
  const float* qw      = (const float*)d_in[2];
  const float* kvw     = (const float*)d_in[3];
  const float* pw      = (const float*)d_in[4];
  const float* pb      = (const float*)d_in[5];
  float* out = (float*)d_out;

  // workspace carve-up (bf16)
  char* ws = (char*)d_ws;
  const size_t xElems   = (size_t)BB * NN * CC;        // 12,582,912
  const size_t qwElems  = (size_t)CC * CC;             //    589,824
  const size_t kvwElems = (size_t)2 * CC * CC;         //  1,179,648
  const size_t qElems   = (size_t)BB * HH * NN * HDD;  // 12,582,912
  const size_t kvElems  = (size_t)BB * HH * NKVV * HDD;//  3,145,728

  bf16_t* xb   = (bf16_t*)ws;                 ws += xElems * 2;
  bf16_t* qwb  = (bf16_t*)ws;                 ws += qwElems * 2;
  bf16_t* kvwb = (bf16_t*)ws;                 ws += kvwElems * 2;
  bf16_t* pwb  = (bf16_t*)ws;                 ws += qwElems * 2;
  bf16_t* Q    = (bf16_t*)ws;                 ws += qElems * 2;
  bf16_t* K    = (bf16_t*)ws;                 ws += kvElems * 2;
  bf16_t* Vt   = (bf16_t*)ws;                 ws += kvElems * 2;
  bf16_t* Ao   = (bf16_t*)ws;

  const dim3 blk(256);
  // one-shot bf16 conversions (each thread converts 8 elements)
  cvt_kernel<<<(int)(xElems   / 2048), blk, 0, stream>>>(x,   xb);
  cvt_kernel<<<(int)(qwElems  / 2048), blk, 0, stream>>>(qw,  qwb);
  cvt_kernel<<<(int)(kvwElems / 2048), blk, 0, stream>>>(kvw, kvwb);
  cvt_kernel<<<(int)(qwElems  / 2048), blk, 0, stream>>>(pw,  pwb);

  qproj_kernel <<<768, blk, 0, stream>>>(xb, qwb, Q);                 // 6144 waves
  kvproj_kernel<<<384, blk, 0, stream>>>(xb, key_ind, kvwb, K, Vt);   // 3072 waves
  attn_kernel  <<<1536, blk, 0, stream>>>(Q, K, Vt, Ao);              // 12288 waves
  oproj_kernel <<<768, blk, 0, stream>>>(Ao, pwb, pb, out);           // 6144 waves
}
